// EncoderRNN_13056700580326
// MI455X (gfx1250) — matrix operand order repaired and checked by reading, not compiled, not run on hardware
//
#include <hip/hip_runtime.h>

// GRU scan, MI455X (gfx1250), wave32 + WMMA bf16.
// T=1024, B=256, H=256.
#define T_STEPS 1024
#define B_SZ    256
#define H_SZ    256

typedef __attribute__((ext_vector_type(16))) __bf16 v16bf;
typedef __attribute__((ext_vector_type(8)))  float  v8f;
typedef __attribute__((ext_vector_type(4)))  float  v4f;

__device__ __forceinline__ unsigned short f2bf(float f) {
    union { float f; unsigned int u; } c; c.f = f;
    unsigned int u = c.u;
    // round-to-nearest-even f32 -> bf16
    unsigned int r = (u + 0x7FFFu + ((u >> 16) & 1u)) >> 16;
    return (unsigned short)r;
}

// ---------------------------------------------------------------------------
// Phase 0: pre-swizzle the six [H,H] f32 weight matrices into bf16 WMMA
// B-fragments. Fragment (g, nt, kt): lane L holds N = nt*16 + (L&15),
// K(e) = kt*32 + (L>>4)*8 + e + (e>=8 ? 8 : 0), e = 0..15 (CDNA5 16-bit B 32x16 layout).
// ws layout: [6][16 nt][8 kt][32 lanes][16 bf16]  -> 768 KB, L2-resident.
// ---------------------------------------------------------------------------
__global__ void gru_prep_weights(const float* __restrict__ Wir,
                                 const float* __restrict__ Wiz,
                                 const float* __restrict__ Win,
                                 const float* __restrict__ Whr,
                                 const float* __restrict__ Whz,
                                 const float* __restrict__ Whn,
                                 unsigned short* __restrict__ wfrag) {
    const int frag = blockIdx.x;            // g*128 + nt*8 + kt, 768 blocks
    const int g  = frag >> 7;
    const int nt = (frag >> 3) & 15;
    const int kt = frag & 7;
    const int L  = threadIdx.x;             // 32 threads = 1 wave
    const float* W;
    switch (g) {
        case 0:  W = Wir; break;
        case 1:  W = Wiz; break;
        case 2:  W = Win; break;
        case 3:  W = Whr; break;
        case 4:  W = Whz; break;
        default: W = Whn; break;
    }
    const int n     = nt * 16 + (L & 15);
    const int kbase = kt * 32 + (L >> 4) * 8;
    unsigned short* out = wfrag + ((size_t)frag * 32 + L) * 16;
#pragma unroll
    for (int e = 0; e < 16; ++e)
        out[e] = f2bf(W[(size_t)(kbase + e + (e >= 8 ? 8 : 0)) * H_SZ + n]);
}

// ---------------------------------------------------------------------------
// Store one f32 hidden value (row, col) into the bf16 A-fragment LDS image.
// CDNA5 16-bit A 16x32 layout: lane = grp*16 + row (grp = ((col&31)>>3)&1),
// element e = ((col>>4)&1)*8 + (col&7) within the 32-wide K tile col>>5.
// ---------------------------------------------------------------------------
__device__ __forceinline__ void store_hA(unsigned short (*hAbuf)[32][16],
                                         int col, int row, unsigned short v) {
    const int kt  = col >> 5;
    const int cc  = col & 31;
    const int grp = (cc >> 3) & 1;
    const int e   = ((cc >> 4) & 1) * 8 + (cc & 7);
    hAbuf[kt][grp * 16 + row][e] = v;
}

// ---------------------------------------------------------------------------
// Phase 1: persistent batch-partitioned GRU scan.
// Grid: 16 WGs (16 batch rows each) x 512 threads (16 waves).
// Each wave owns ONE 16-column tile for the whole run.
//  - hA: hidden state as bf16 A-frags, pre-masked with resets[t], ping-pong.
//  - xA: x_t as bf16 A-frags, staged one step ahead by waves 0..7 (software
//    pipeline: global loads for t+1 overlap step-t WMMAs; GEMM phase is
//    LDS-only on the critical path). One barrier per step covers both.
// ---------------------------------------------------------------------------
__global__ void __launch_bounds__(512, 1)
gru_scan(const float* __restrict__ xs,
         const unsigned char* __restrict__ resets,
         const float* __restrict__ h0,
         const float* __restrict__ b_ir, const float* __restrict__ b_iz,
         const float* __restrict__ b_in, const float* __restrict__ b_hn,
         const unsigned short* __restrict__ wfrag,
         float* __restrict__ out) {
    __shared__ unsigned short hA[2][8][32][16];   // ping-pong h A-frags, 16 KB
    __shared__ unsigned short xA[2][8][32][16];   // ping-pong x A-frags, 16 KB

    const int tid     = threadIdx.x;
    const int L       = tid & 31;
    const int nt      = tid >> 5;         // wave id == column tile id (0..15)
    const int lcol    = L & 15;
    const int halfsel = L >> 4;
    const int b0      = blockIdx.x * 16;
    const int col     = nt * 16 + lcol;
    const bool producer = (nt < 8);       // waves 0..7 stage x frag kt = nt

    float* __restrict__ ys = out + (size_t)B_SZ * H_SZ;   // out = [h_final | ys]
    const v16bf* __restrict__ wv = (const v16bf*)wfrag;
    const size_t gstride = (size_t)16 * 8 * 32;           // v16bf frags per gate

    // Biases (column-indexed, broadcast over the 8 C rows).
    const float biasR  = b_ir[col];
    const float biasZ  = b_iz[col];
    const float biasNX = b_in[col];
    const float biasNH = b_hn[col];

    // Register-resident hidden-side weights for the r and z gates:
    // 2 gates x 8 kt x v16bf = 128 VGPRs, loop-invariant.
    v16bf whrR[8], whzR[8];
#pragma unroll
    for (int kt = 0; kt < 8; ++kt) {
        const size_t base = (((size_t)nt) * 8 + kt) * 32 + L;
        whrR[kt] = wv[3 * gstride + base];
        whzR[kt] = wv[4 * gstride + base];
    }

    // Producer-lane x addressing: frag kt=nt, row = L&15, K base = halfsel*8.
    const size_t xlane_off = ((size_t)b0 + lcol) * H_SZ + nt * 32 + halfsel * 8;

    // Prologue: stage x for t=0; init hidden state (hA masked by resets[0]).
    if (producer) {
        const float* xr = xs + xlane_off;
        v4f p0 = *(const v4f*)(xr + 0);
        v4f p1 = *(const v4f*)(xr + 4);
        v4f p2 = *(const v4f*)(xr + 16);
        v4f p3 = *(const v4f*)(xr + 20);
        v16bf xa;
#pragma unroll
        for (int e = 0; e < 4; ++e) {
            xa[e]      = (__bf16)p0[e];
            xa[4 + e]  = (__bf16)p1[e];
            xa[8 + e]  = (__bf16)p2[e];
            xa[12 + e] = (__bf16)p3[e];
        }
        *(v16bf*)&xA[0][nt][L][0] = xa;
    }
    float hprev[8];
#pragma unroll
    for (int v = 0; v < 8; ++v) {
        const int row = halfsel * 8 + v;
        const float h = h0[(size_t)(b0 + row) * H_SZ + col];
        hprev[v] = h;
        const float hm = (resets[(size_t)0 * B_SZ + b0 + row] != 0) ? 0.0f : h;
        store_hA(hA[0], col, row, f2bf(hm));
    }
    __syncthreads();

    for (int t = 0; t < T_STEPS; ++t) {
        // Keep streamed weight loads inside the loop (L2 hits), not hoisted.
        asm volatile("" ::: "memory");

        const int buf = t & 1;
        const unsigned char* __restrict__ rrow = resets + (size_t)t * B_SZ + b0;

        // Issue next step's x loads early; they drain while WMMAs run.
        v4f p0, p1, p2, p3;
        const bool stage = producer && (t + 1 < T_STEPS);
        if (stage) {
            const float* xr = xs + ((size_t)(t + 1) * B_SZ) * H_SZ + xlane_off;
            p0 = *(const v4f*)(xr + 0);
            p1 = *(const v4f*)(xr + 4);
            p2 = *(const v4f*)(xr + 16);
            p3 = *(const v4f*)(xr + 20);
        }

        // Accumulators, initialized with fused biases.
        v8f accR, accZ, accNX, accNH;
#pragma unroll
        for (int v = 0; v < 8; ++v) {
            accR[v]  = biasR;
            accZ[v]  = biasZ;
            accNX[v] = biasNX;
            accNH[v] = biasNH;
        }

        // GEMM phase: LDS-only A operands, mask-free (hA is pre-masked).
#pragma unroll
        for (int kt = 0; kt < 8; ++kt) {
            v16bf xa = *(const v16bf*)&xA[buf][kt][L][0];
            v16bf ha = *(const v16bf*)&hA[buf][kt][L][0];

            const size_t base = (((size_t)nt) * 8 + kt) * 32 + L;
            v16bf wir = wv[0 * gstride + base];
            v16bf wiz = wv[1 * gstride + base];
            v16bf win = wv[2 * gstride + base];
            v16bf whn = wv[5 * gstride + base];

            // Dependent WMMA pairs are >= 3 apart => co-exec hides hazards.
            accR  = __builtin_amdgcn_wmma_f32_16x16x32_bf16(
                        false, xa, false, wir,      (short)0, accR,  false, false);
            accZ  = __builtin_amdgcn_wmma_f32_16x16x32_bf16(
                        false, xa, false, wiz,      (short)0, accZ,  false, false);
            accNX = __builtin_amdgcn_wmma_f32_16x16x32_bf16(
                        false, xa, false, win,      (short)0, accNX, false, false);
            accR  = __builtin_amdgcn_wmma_f32_16x16x32_bf16(
                        false, ha, false, whrR[kt], (short)0, accR,  false, false);
            accZ  = __builtin_amdgcn_wmma_f32_16x16x32_bf16(
                        false, ha, false, whzR[kt], (short)0, accZ,  false, false);
            accNH = __builtin_amdgcn_wmma_f32_16x16x32_bf16(
                        false, ha, false, whn,      (short)0, accNH, false, false);
        }

        // Gate epilogue + state update (C/D layout: row = halfsel*8+v, N = lcol).
        // hA image for t+1 is stored pre-masked with resets[t+1].
        const unsigned char* __restrict__ rnext =
            resets + (size_t)(t + 1 < T_STEPS ? t + 1 : t) * B_SZ + b0;
        const bool has_next = (t + 1 < T_STEPS);
#pragma unroll
        for (int v = 0; v < 8; ++v) {
            const int row = halfsel * 8 + v;
            const float hp = (rrow[row] != 0) ? 0.0f : hprev[v];
            const float r  = 1.0f / (1.0f + __expf(-accR[v]));
            const float z  = 1.0f / (1.0f + __expf(-accZ[v]));
            const float a  = accNX[v] + r * accNH[v];
            const float e2 = __expf(-2.0f * a);
            const float n  = (1.0f - e2) / (1.0f + e2);
            const float hn = (1.0f - z) * n + z * hp;
            hprev[v] = hn;
            ys[((size_t)t * B_SZ + b0 + row) * H_SZ + col] = hn;
            const float hm = (has_next && rnext[row] != 0) ? 0.0f : hn;
            store_hA(hA[buf ^ 1], col, row, f2bf(hm));
        }

        // Finish staging x for t+1 (convert + LDS publish).
        if (stage) {
            v16bf xa;
#pragma unroll
            for (int e = 0; e < 4; ++e) {
                xa[e]      = (__bf16)p0[e];
                xa[4 + e]  = (__bf16)p1[e];
                xa[8 + e]  = (__bf16)p2[e];
                xa[12 + e] = (__bf16)p3[e];
            }
            *(v16bf*)&xA[buf ^ 1][nt][L][0] = xa;
        }
        __syncthreads();   // publish next step's h/x A-frags (ping-pong)
    }

    // Final carry h_T -> out[0 : B*H)
#pragma unroll
    for (int v = 0; v < 8; ++v) {
        const int row = halfsel * 8 + v;
        out[(size_t)(b0 + row) * H_SZ + col] = hprev[v];
    }
}

// ---------------------------------------------------------------------------
extern "C" void kernel_launch(void* const* d_in, const int* in_sizes, int n_in,
                              void* d_out, int out_size, void* d_ws, size_t ws_size,
                              hipStream_t stream) {
    const float*         xs     = (const float*)d_in[0];
    const unsigned char* resets = (const unsigned char*)d_in[1];
    const float*         h0     = (const float*)d_in[2];
    const float*         Wir    = (const float*)d_in[3];
    const float*         Wiz    = (const float*)d_in[4];
    const float*         Win    = (const float*)d_in[5];
    const float*         b_ir   = (const float*)d_in[6];
    const float*         b_iz   = (const float*)d_in[7];
    const float*         b_in   = (const float*)d_in[8];
    const float*         Whr    = (const float*)d_in[9];
    const float*         Whz    = (const float*)d_in[10];
    const float*         Whn    = (const float*)d_in[11];
    const float*         b_hn   = (const float*)d_in[12];

    unsigned short* wfrag = (unsigned short*)d_ws;   // needs 768 KB
    float* out = (float*)d_out;

    // Phase 0: weight pre-swizzle to bf16 WMMA fragments (6*16*8 frags).
    gru_prep_weights<<<768, 32, 0, stream>>>(Wir, Wiz, Win, Whr, Whz, Whn, wfrag);

    // Phase 1: batch-partitioned persistent scan: 16 WGs x 16 waves.
    gru_scan<<<16, 512, 0, stream>>>(xs, resets, h0,
                                     b_ir, b_iz, b_in, b_hn,
                                     wfrag, out);
}
